// DRHGTModel_81097572483647
// MI455X (gfx1250) — compile-verified, matrix operand order checked by reading, not run on hardware
//
#include <hip/hip_runtime.h>

// ---------------------------------------------------------------------------
// MI455X (gfx1250) implementation of the DR-HGT model.
//  * All dense GEMMs run on v_wmma_f32_16x16x32_bf16 (bf16 in, f32 acc).
//  * Per-edge relation GEMMs hoisted to node level ((k@Watt)[src]).
//  * Operands zero-padded to K%32==0 and weights pre-transposed on device so
//    every WMMA fragment is exactly two aligned global_load_b128 per lane —
//    no per-element predication in the hot loop.
//  * MLP layer-1 gathers its A rows from L2-resident drug/dis feature tables
//    inside the fragment loader (no 400MB gathered intermediate).
//  * Edge softmax / scatter / layernorm / BN are wave32-shaped VALU kernels.
// ---------------------------------------------------------------------------

typedef __bf16 bf16;
typedef __attribute__((ext_vector_type(16))) __bf16 v16bf;
typedef __attribute__((ext_vector_type(8)))  float  v8f;

#define NP_  20000
#define ND_  2000
#define NG_  2000
#define E_DP 150000
#define E_GP 150000
#define E_T  100000
#define PDIM 400
#define DDIM 300
#define GDIM 300
#define PPAD 416   // 400 -> %32
#define DPAD 320   // 300 -> %32
#define DH   128
#define LAY  4

union FragU  { v16bf v; uint4  u[2]; };

// ---------------------------------------------------------------------------
// Fragment loaders: CDNA5 16x16x32 bf16 layout, wave32.
// A (16x32): lane holds row m=lane&15; K-half selected by lane>=16 (+8),
//   元素 pairs at kb+{0,2,4,6} and kb+{16,18,20,22} -> two contiguous 8-elem
//   runs -> two 16B loads.  B via pre-transposed Bt[N x Kpad]: identical
//   addressing with row n.  C/D: VGPR r -> row r + 8*(lane>=16), col lane&15.
// ---------------------------------------------------------------------------
__device__ inline v16bf load_frag16(const bf16* __restrict__ base, long ld,
                                    int row, int kb) {
  const bf16* p = base + (long)row * ld + kb;
  FragU f;
  f.u[0] = *(const uint4*)(p);        // elements kb .. kb+7
  f.u[1] = *(const uint4*)(p + 16);   // elements kb+16 .. kb+23
  return f.v;
}

__device__ inline void store_tile(void* __restrict__ C, long ldc,
                                  int m0, int n0, v8f acc,
                                  const float* __restrict__ bias, int store_bf) {
  int lane  = threadIdx.x & 31;
  int col   = n0 + (lane & 15);
  int rbase = m0 + ((lane & 16) ? 8 : 0);
  float bv = bias ? bias[col] : 0.0f;
#pragma unroll
  for (int r = 0; r < 8; ++r) {
    int row = rbase + r;
    float v = acc[r] + bv;
    if (store_bf) ((bf16*)C)[(long)row * ldc + col] = (bf16)v;
    else          ((float*)C)[(long)row * ldc + col] = v;
  }
}

// ---------------------------------------------------------------------------
// GEMM: C[MxN] = A[MxKpad](bf16) * Bt[NxKpad](bf16, pre-transposed) + bias
// block = 128 threads (4 waves); tile 16(M) x 128(N); grid = (M/16, N/128).
// Kpad % 32 == 0; M % 16 == 0; N % 128 == 0  ->  zero guards in the loop.
// ---------------------------------------------------------------------------
__global__ void gemm_bf16_kernel(const bf16* __restrict__ A, long lda,
                                 const bf16* __restrict__ Bt, long ldbt,
                                 void* __restrict__ C, long ldc,
                                 int Kpad, const float* __restrict__ bias,
                                 int store_bf) {
  int lane = threadIdx.x & 31;
  int wave = threadIdx.x >> 5;
  int m0 = blockIdx.x * 16;
  int n0 = blockIdx.y * 128 + wave * 32;
  int arow = m0 + (lane & 15);
  int brow = n0 + (lane & 15);
  int koff = (lane & 16) ? 8 : 0;
  v8f acc0 = {};
  v8f acc1 = {};
  for (int k0 = 0; k0 < Kpad; k0 += 32) {
    int kb = k0 + koff;
    v16bf a  = load_frag16(A,  lda,  arow, kb);
    v16bf b0 = load_frag16(Bt, ldbt, brow, kb);
    v16bf b1 = load_frag16(Bt, ldbt, brow + 16, kb);
    __builtin_prefetch(A + (long)arow * lda + kb + 32, 0, 1);
    acc0 = __builtin_amdgcn_wmma_f32_16x16x32_bf16(false, a, false, b0,
                                                   (short)0, acc0, false, false);
    acc1 = __builtin_amdgcn_wmma_f32_16x16x32_bf16(false, a, false, b1,
                                                   (short)0, acc1, false, false);
  }
  store_tile(C, ldc, m0, n0,      acc0, bias, store_bf);
  store_tile(C, ldc, m0, n0 + 16, acc1, bias, store_bf);
}

// MLP layer 1 with fused edge gather:
//   X1[e,:] = concat(drug_cat[ts[e]], dis_cat[td[e]]) @ W1 + b1   (bf16 out)
// A rows gathered as aligned float4 from the 8MB L2-resident cat tables.
__global__ void gemm_gather_kernel(const float* __restrict__ dc,
                                   const float* __restrict__ gc,
                                   const int* __restrict__ ts,
                                   const int* __restrict__ td,
                                   const bf16* __restrict__ Bt,  // [512 x 1024]
                                   bf16* __restrict__ C,
                                   const float* __restrict__ bias) {
  int lane = threadIdx.x & 31;
  int wave = threadIdx.x >> 5;
  int m0 = blockIdx.x * 16;
  int n0 = blockIdx.y * 128 + wave * 32;
  int m  = m0 + (lane & 15);
  int brow = n0 + (lane & 15);
  int koff = (lane & 16) ? 8 : 0;
  long srow = (long)ts[m] * 512;
  long grow = (long)td[m] * 512;
  v8f acc0 = {};
  v8f acc1 = {};
  for (int k0 = 0; k0 < 1024; k0 += 32) {
    // K-chunks are 32-aligned -> never straddle the 512 boundary
    const float* row = (k0 < 512) ? (dc + srow) : (gc + grow - 512);
    int kb = k0 + koff;
    float4 x0 = *(const float4*)(row + kb);
    float4 x1 = *(const float4*)(row + kb + 4);
    float4 y0 = *(const float4*)(row + kb + 16);
    float4 y1 = *(const float4*)(row + kb + 20);
    v16bf a;
    a[0]=(bf16)x0.x; a[1]=(bf16)x0.y; a[2]=(bf16)x0.z; a[3]=(bf16)x0.w;
    a[4]=(bf16)x1.x; a[5]=(bf16)x1.y; a[6]=(bf16)x1.z; a[7]=(bf16)x1.w;
    a[8]=(bf16)y0.x; a[9]=(bf16)y0.y; a[10]=(bf16)y0.z; a[11]=(bf16)y0.w;
    a[12]=(bf16)y1.x; a[13]=(bf16)y1.y; a[14]=(bf16)y1.z; a[15]=(bf16)y1.w;
    v16bf b0 = load_frag16(Bt, 1024, brow, kb);
    v16bf b1 = load_frag16(Bt, 1024, brow + 16, kb);
    acc0 = __builtin_amdgcn_wmma_f32_16x16x32_bf16(false, a, false, b0,
                                                   (short)0, acc0, false, false);
    acc1 = __builtin_amdgcn_wmma_f32_16x16x32_bf16(false, a, false, b1,
                                                   (short)0, acc1, false, false);
  }
  store_tile(C, 512, m0, n0,      acc0, bias, 1);
  store_tile(C, 512, m0, n0 + 16, acc1, bias, 1);
}

// ---------------------------------------------------------------------------
// Staging / elementwise kernels
// ---------------------------------------------------------------------------

// f32 [rows x din] -> bf16 [rows x dpad], zero-padded columns
__global__ void cvt_pad_kernel(const float* __restrict__ in,
                               bf16* __restrict__ out,
                               long rows, int din, int dpad) {
  long i = (long)blockIdx.x * blockDim.x + threadIdx.x;
  if (i >= rows * (long)dpad) return;
  long r = i / dpad;
  int  c = (int)(i % dpad);
  out[i] = (c < din) ? (bf16)in[r * din + c] : (bf16)0.0f;
}

// weight transpose+pad: f32 W[K x N] -> bf16 Wt[N x Kpad] (zero-padded K rows)
__global__ void cvt_t_kernel(const float* __restrict__ in,
                             bf16* __restrict__ out,
                             int K, int N, int Kpad) {
  long i = (long)blockIdx.x * blockDim.x + threadIdx.x;
  if (i >= (long)N * Kpad) return;
  int n = (int)(i / Kpad);
  int k = (int)(i % Kpad);
  out[i] = (k < K) ? (bf16)in[(long)k * N + n] : (bf16)0.0f;
}

__global__ void fill_kernel(unsigned* __restrict__ p, unsigned v, long n) {
  long i = (long)blockIdx.x * blockDim.x + threadIdx.x;
  if (i < n) p[i] = v;
}

// wave-per-edge: score = dot(kW[src], q[dst]) * pri / sqrt(128); seg atomic max.
__global__ void score_kernel(const float* __restrict__ kW,
                             const float* __restrict__ q,
                             const int* __restrict__ src,
                             const int* __restrict__ dst,
                             const float* __restrict__ pri, int E,
                             float* __restrict__ score,
                             unsigned* __restrict__ segmax) {
  int e = (int)(((long)blockIdx.x * blockDim.x + threadIdx.x) >> 5);
  int lane = threadIdx.x & 31;
  if (e >= E) return;
  int s = src[e], d = dst[e];
  float acc = 0.0f;
#pragma unroll
  for (int i = 0; i < 4; ++i) {
    int f = lane + i * 32;
    acc += kW[(long)s * DH + f] * q[(long)d * DH + f];
  }
  for (int off = 16; off; off >>= 1) acc += __shfl_down(acc, off, 32);
  if (lane == 0) {
    float sc = acc * pri[0] * 0.08838834764831845f;  // 1/sqrt(128)
    score[e] = sc;
    unsigned u = __float_as_uint(sc);
    u = (u & 0x80000000u) ? ~u : (u | 0x80000000u);  // order-preserving encode
    atomicMax(segmax + d, u);
  }
}

__global__ void expsum_kernel(const float* __restrict__ score,
                              const int* __restrict__ dst,
                              const unsigned* __restrict__ segmax,
                              float* __restrict__ att,
                              float* __restrict__ segs, int E) {
  long i = (long)blockIdx.x * blockDim.x + threadIdx.x;
  if (i >= E) return;
  int d = dst[i];
  unsigned u = segmax[d];
  unsigned v = (u & 0x80000000u) ? (u & 0x7fffffffu) : ~u;  // decode
  float m = __uint_as_float(v);
  float ex = __expf(score[i] - m);
  att[i] = ex;
  atomicAdd(segs + d, ex);
}

// wave-per-edge: agg[dst] += (att/sum) * vW[src]
__global__ void scatter_kernel(const float* __restrict__ att,
                               const float* __restrict__ segs,
                               const float* __restrict__ vW,
                               const int* __restrict__ src,
                               const int* __restrict__ dst,
                               float* __restrict__ agg, int E) {
  int e = (int)(((long)blockIdx.x * blockDim.x + threadIdx.x) >> 5);
  int lane = threadIdx.x & 31;
  if (e >= E) return;
  int s = src[e], d = dst[e];
  float w = att[e] / (segs[d] + 1e-9f);
#pragma unroll
  for (int i = 0; i < 4; ++i) {
    int f = lane + i * 32;
    atomicAdd(agg + (long)d * DH + f, w * vW[(long)s * DH + f]);
  }
}

// wave-per-row: out = LN(alpha*trans + (1-alpha)*prev) * g + b, optional cat copy
__global__ void mix_ln_kernel(const float* __restrict__ trans,
                              const float* __restrict__ prev,
                              const float* __restrict__ skip,
                              const float* __restrict__ g,
                              const float* __restrict__ b,
                              float* __restrict__ out,
                              float* __restrict__ cat, long catld, long catoff,
                              int N) {
  int row = (int)(((long)blockIdx.x * blockDim.x + threadIdx.x) >> 5);
  int lane = threadIdx.x & 31;
  if (row >= N) return;
  float alpha = prev ? (1.0f / (1.0f + __expf(-skip[0]))) : 1.0f;
  float vals[4];
  float mean = 0.0f;
#pragma unroll
  for (int i = 0; i < 4; ++i) {
    int f = lane + i * 32;
    float t = trans[(long)row * DH + f];
    float v = prev ? (alpha * t + (1.0f - alpha) * prev[(long)row * DH + f]) : t;
    vals[i] = v;
    mean += v;
  }
  for (int off = 16; off; off >>= 1) mean += __shfl_down(mean, off, 32);
  mean = __shfl(mean, 0, 32) * (1.0f / 128.0f);
  float var = 0.0f;
#pragma unroll
  for (int i = 0; i < 4; ++i) { float d2 = vals[i] - mean; var += d2 * d2; }
  for (int off = 16; off; off >>= 1) var += __shfl_down(var, off, 32);
  var = __shfl(var, 0, 32) * (1.0f / 128.0f);
  float inv = rsqrtf(var + 1e-5f);
#pragma unroll
  for (int i = 0; i < 4; ++i) {
    int f = lane + i * 32;
    float o = (vals[i] - mean) * inv * g[f] + b[f];
    out[(long)row * DH + f] = o;
    if (cat) cat[(long)row * catld + catoff + f] = o;
  }
}

// wave-per-row: ProteinFusionAttention over 2 sources
__global__ void fuse_kernel(const float* __restrict__ pfd,
                            const float* __restrict__ pfs,
                            float* __restrict__ hp, int N) {
  int row = (int)(((long)blockIdx.x * blockDim.x + threadIdx.x) >> 5);
  int lane = threadIdx.x & 31;
  if (row >= N) return;
  float m0 = -3.4e38f, m1 = -3.4e38f;
#pragma unroll
  for (int i = 0; i < 4; ++i) {
    int f = lane + i * 32;
    m0 = fmaxf(m0, pfd[(long)row * DH + f]);
    m1 = fmaxf(m1, pfs[(long)row * DH + f]);
  }
  for (int off = 16; off; off >>= 1) {
    m0 = fmaxf(m0, __shfl_down(m0, off, 32));
    m1 = fmaxf(m1, __shfl_down(m1, off, 32));
  }
  m0 = __shfl(m0, 0, 32);
  m1 = __shfl(m1, 0, 32);
  float mm = fmaxf(m0, m1);
  float a0 = __expf(m0 - mm), a1 = __expf(m1 - mm);
  float s = a0 + a1;
  a0 /= s; a1 /= s;
#pragma unroll
  for (int i = 0; i < 4; ++i) {
    int f = lane + i * 32;
    hp[(long)row * DH + f] =
        a0 * pfd[(long)row * DH + f] + a1 * pfs[(long)row * DH + f];
  }
}

// block-per-column batchnorm stats over bf16 matrix
__global__ void bn_stats_kernel(const bf16* __restrict__ x, int M, int N,
                                float* __restrict__ mean,
                                float* __restrict__ invstd) {
  int col = blockIdx.x;
  float s = 0.0f, ss = 0.0f;
  for (int r = threadIdx.x; r < M; r += blockDim.x) {
    float v = (float)x[(long)r * N + col];
    s += v; ss += v * v;
  }
  __shared__ float sh[256], sh2[256];
  sh[threadIdx.x] = s; sh2[threadIdx.x] = ss;
  __syncthreads();
  for (int o = 128; o; o >>= 1) {
    if ((int)threadIdx.x < o) {
      sh[threadIdx.x]  += sh[threadIdx.x + o];
      sh2[threadIdx.x] += sh2[threadIdx.x + o];
    }
    __syncthreads();
  }
  if (threadIdx.x == 0) {
    float mu  = sh[0] / (float)M;
    float var = sh2[0] / (float)M - mu * mu;
    mean[col]   = mu;
    invstd[col] = rsqrtf(var + 1e-5f);
  }
}

__global__ void bn_relu_kernel(bf16* __restrict__ x, long total, int N,
                               const float* __restrict__ mean,
                               const float* __restrict__ invstd,
                               const float* __restrict__ g,
                               const float* __restrict__ b) {
  long i = (long)blockIdx.x * blockDim.x + threadIdx.x;
  if (i >= total) return;
  int col = (int)(i % N);
  float v = ((float)x[i] - mean[col]) * invstd[col] * g[col] + b[col];
  x[i] = (bf16)fmaxf(v, 0.0f);
}

// wave-per-row final: sigmoid(x2 @ W3 + b3)
__global__ void final_kernel(const bf16* __restrict__ x2,
                             const float* __restrict__ W3,
                             const float* __restrict__ b3,
                             float* __restrict__ out, int M) {
  int row = (int)(((long)blockIdx.x * blockDim.x + threadIdx.x) >> 5);
  int lane = threadIdx.x & 31;
  if (row >= M) return;
  float acc = 0.0f;
#pragma unroll
  for (int i = 0; i < 8; ++i) {
    int k = lane + i * 32;
    acc += (float)x2[(long)row * 256 + k] * W3[k];
  }
  for (int off = 16; off; off >>= 1) acc += __shfl_down(acc, off, 32);
  if (lane == 0) out[row] = 1.0f / (1.0f + __expf(-(acc + b3[0])));
}

// ---------------------------------------------------------------------------
// Host-side orchestration
// ---------------------------------------------------------------------------
namespace {

struct HgtScratch {
  bf16*  wbuf;           // transposed bf16 weight staging (max 512*1024)
  float *kf, *qf, *vf;   // [NP_*DH]
  bf16*  kvbf;           // [NP_*DH]
  float *kW, *vW;        // [NP_*DH]
  float *score, *att;    // [E_DP]
  unsigned* segmax;      // [NP_]
  float* segs;           // [NP_]
  float* agg;            // [NP_*DH]
  bf16*  aggbf;          // [NP_*DH]
  float* trans;          // [NP_*DH]
};

inline void cvt_pad(hipStream_t st, const float* in, bf16* out,
                    long rows, int din, int dpad) {
  long n = rows * (long)dpad;
  cvt_pad_kernel<<<dim3((unsigned)((n + 255) / 256)), 256, 0, st>>>(
      in, out, rows, din, dpad);
}

inline void cvt_t(hipStream_t st, const float* in, bf16* out,
                  int K, int N, int Kpad) {
  long n = (long)N * Kpad;
  cvt_t_kernel<<<dim3((unsigned)((n + 255) / 256)), 256, 0, st>>>(
      in, out, K, N, Kpad);
}

inline void fill(hipStream_t st, void* p, unsigned v, long n) {
  fill_kernel<<<dim3((unsigned)((n + 255) / 256)), 256, 0, st>>>((unsigned*)p, v, n);
}

// M%16==0, N%128==0, Kpad%32==0
inline void gemm(hipStream_t st, const bf16* A, long lda, const bf16* Bt,
                 long ldbt, void* C, long ldc, int M, int N, int Kpad,
                 const float* bias, int store_bf) {
  dim3 grid((unsigned)(M / 16), (unsigned)(N / 128));
  gemm_bf16_kernel<<<grid, 128, 0, st>>>(A, lda, Bt, ldbt, C, ldc, Kpad,
                                         bias, store_bf);
}

// One HGT block. P[] layout: 0 Wk,1 bk,2 Wq,3 bq,4 Wv,5 bv,6 Watt,7 Wmsg,
//                            8 Wa,9 ba,10 pri,11 skip,12 g,13 b
void run_hgt(hipStream_t st, HgtScratch& S,
             const bf16* hs_bf, int Ns, int ds, int dspad,
             const bf16* hd_bf, int Nd, int dd, int ddpad,
             const float* prev_dst,
             const int* esrc, const int* edst, int E,
             const float* const* P,
             float* out, float* cat, long catld, long catoff) {
  // k = hs @ Wk + bk ; q = hd @ Wq + bq ; v = hs @ Wv + bv
  cvt_t(st, P[0], S.wbuf, ds, DH, dspad);
  gemm(st, hs_bf, dspad, S.wbuf, dspad, S.kf, DH, Ns, DH, dspad, P[1], 0);
  cvt_t(st, P[2], S.wbuf, dd, DH, ddpad);
  gemm(st, hd_bf, ddpad, S.wbuf, ddpad, S.qf, DH, Nd, DH, ddpad, P[3], 0);
  cvt_t(st, P[4], S.wbuf, ds, DH, dspad);
  gemm(st, hs_bf, dspad, S.wbuf, dspad, S.vf, DH, Ns, DH, dspad, P[5], 0);
  // hoisted relation transforms: kW = k@Watt, vW = v@Wmsg  (node level)
  cvt_pad(st, S.kf, S.kvbf, Ns, DH, DH);
  cvt_t(st, P[6], S.wbuf, DH, DH, DH);
  gemm(st, S.kvbf, DH, S.wbuf, DH, S.kW, DH, Ns, DH, DH, nullptr, 0);
  cvt_pad(st, S.vf, S.kvbf, Ns, DH, DH);
  cvt_t(st, P[7], S.wbuf, DH, DH, DH);
  gemm(st, S.kvbf, DH, S.wbuf, DH, S.vW, DH, Ns, DH, DH, nullptr, 0);
  // segment softmax + scatter
  fill(st, S.segmax, 0x007fffffu, Nd);       // encode(-inf)
  fill(st, S.segs, 0u, Nd);
  fill(st, S.agg, 0u, (long)Nd * DH);
  dim3 eg((unsigned)((E + 3) / 4));
  score_kernel<<<eg, 128, 0, st>>>(S.kW, S.qf, esrc, edst, P[10], E,
                                   S.score, S.segmax);
  expsum_kernel<<<dim3((unsigned)((E + 255) / 256)), 256, 0, st>>>(
      S.score, edst, S.segmax, S.att, S.segs, E);
  scatter_kernel<<<eg, 128, 0, st>>>(S.att, S.segs, S.vW, esrc, edst, S.agg, E);
  // trans = agg @ Wa + ba ; skip-mix ; layernorm
  cvt_pad(st, S.agg, S.aggbf, Nd, DH, DH);
  cvt_t(st, P[8], S.wbuf, DH, DH, DH);
  gemm(st, S.aggbf, DH, S.wbuf, DH, S.trans, DH, Nd, DH, DH, P[9], 0);
  mix_ln_kernel<<<dim3((unsigned)((Nd + 3) / 4)), 128, 0, st>>>(
      S.trans, prev_dst, P[11], P[12], P[13], out, cat, catld, catoff, Nd);
}

}  // namespace

extern "C" void kernel_launch(void* const* d_in, const int* in_sizes, int n_in,
                              void* d_out, int out_size, void* d_ws, size_t ws_size,
                              hipStream_t stream) {
  (void)in_sizes; (void)n_in; (void)out_size; (void)ws_size;
  const float* h_protein = (const float*)d_in[0];
  const float* h_drug    = (const float*)d_in[1];
  const float* h_disease = (const float*)d_in[2];
  const int* dp_src = (const int*)d_in[3];
  const int* dp_dst = (const int*)d_in[4];
  const int* gp_src = (const int*)d_in[5];
  const int* gp_dst = (const int*)d_in[6];
  const int* t_src  = (const int*)d_in[7];
  const int* t_dst  = (const int*)d_in[8];

  // params flattened in dict-insertion order:
  // layers[i] -> {DT,DG,TD,GD} each 14 leaves, then mlp (10 leaves)
  auto PRM = [&](int l, int blk, int idx) -> const float* {
    return (const float*)d_in[9 + (l * 4 + blk) * 14 + idx];
  };
  const int MLPB = 9 + 16 * 14;  // 233
  const float* W1  = (const float*)d_in[MLPB + 0];
  const float* b1  = (const float*)d_in[MLPB + 1];
  const float* g1  = (const float*)d_in[MLPB + 2];
  const float* bn1 = (const float*)d_in[MLPB + 3];
  const float* W2  = (const float*)d_in[MLPB + 4];
  const float* b2  = (const float*)d_in[MLPB + 5];
  const float* g2  = (const float*)d_in[MLPB + 6];
  const float* bn2 = (const float*)d_in[MLPB + 7];
  const float* W3  = (const float*)d_in[MLPB + 8];
  const float* b3  = (const float*)d_in[MLPB + 9];

  // ---- bump allocator over d_ws ----
  char* wsp = (char*)d_ws;
  size_t off = 0;
  auto alloc = [&](size_t bytes) -> void* {
    off = (off + 255) & ~(size_t)255;
    void* p = wsp + off;
    off += bytes;
    return p;
  };

  float* hp  = (float*)alloc((size_t)NP_ * DH * 4);
  float* hd  = (float*)alloc((size_t)ND_ * DH * 4);
  float* hg  = (float*)alloc((size_t)NG_ * DH * 4);
  bf16*  hpbf = (bf16*)alloc((size_t)NP_ * PPAD * 2);
  bf16*  hdbf = (bf16*)alloc((size_t)ND_ * DPAD * 2);
  bf16*  hgbf = (bf16*)alloc((size_t)NG_ * DPAD * 2);
  float* pfd = (float*)alloc((size_t)NP_ * DH * 4);
  float* pfs = (float*)alloc((size_t)NP_ * DH * 4);
  float* drugcat = (float*)alloc((size_t)ND_ * 512 * 4);
  float* discat  = (float*)alloc((size_t)NG_ * 512 * 4);

  HgtScratch S;
  S.wbuf   = (bf16*)alloc((size_t)512 * 1024 * 2);
  S.kf     = (float*)alloc((size_t)NP_ * DH * 4);
  S.qf     = (float*)alloc((size_t)NP_ * DH * 4);
  S.vf     = (float*)alloc((size_t)NP_ * DH * 4);
  S.kvbf   = (bf16*)alloc((size_t)NP_ * DH * 2);
  S.kW     = (float*)alloc((size_t)NP_ * DH * 4);
  S.vW     = (float*)alloc((size_t)NP_ * DH * 4);
  S.score  = (float*)alloc((size_t)E_DP * 4);
  S.att    = (float*)alloc((size_t)E_DP * 4);
  S.segmax = (unsigned*)alloc((size_t)NP_ * 4);
  S.segs   = (float*)alloc((size_t)NP_ * 4);
  S.agg    = (float*)alloc((size_t)NP_ * DH * 4);
  S.aggbf  = (bf16*)alloc((size_t)NP_ * DH * 2);
  S.trans  = (float*)alloc((size_t)NP_ * DH * 4);

  bf16* x1 = (bf16*)alloc((size_t)E_T * 512 * 2);
  bf16* x2 = (bf16*)alloc((size_t)E_T * 256 * 2);
  float* bmean   = (float*)alloc(512 * 4);
  float* binvstd = (float*)alloc(512 * 4);

  const float* hp_cur = h_protein; int dp = PDIM, dppad = PPAD;
  const float* hd_cur = h_drug;    int dd = DDIM, ddpad = DPAD;
  const float* hg_cur = h_disease; int dg = GDIM, dgpad = DPAD;

  for (int l = 0; l < LAY; ++l) {
    cvt_pad(stream, hp_cur, hpbf, NP_, dp, dppad);
    cvt_pad(stream, hd_cur, hdbf, ND_, dd, ddpad);
    cvt_pad(stream, hg_cur, hgbf, NG_, dg, dgpad);

    const float* P[14];

    // DT: drug -> protein
    for (int t = 0; t < 14; ++t) P[t] = PRM(l, 0, t);
    run_hgt(stream, S, hdbf, ND_, dd, ddpad, hpbf, NP_, dp, dppad,
            (dp == DH) ? hp_cur : nullptr,
            dp_src, dp_dst, E_DP, P, pfd, nullptr, 0, 0);

    // DG: disease -> protein
    for (int t = 0; t < 14; ++t) P[t] = PRM(l, 1, t);
    run_hgt(stream, S, hgbf, NG_, dg, dgpad, hpbf, NP_, dp, dppad,
            (dp == DH) ? hp_cur : nullptr,
            gp_src, gp_dst, E_GP, P, pfs, nullptr, 0, 0);

    // TD: protein -> drug (edges reversed)
    for (int t = 0; t < 14; ++t) P[t] = PRM(l, 2, t);
    run_hgt(stream, S, hpbf, NP_, dp, dppad, hdbf, ND_, dd, ddpad,
            (dd == DH) ? hd_cur : nullptr,
            dp_dst, dp_src, E_DP, P, hd, drugcat, 512, (long)l * DH);

    // GD: protein -> disease (edges reversed)
    for (int t = 0; t < 14; ++t) P[t] = PRM(l, 3, t);
    run_hgt(stream, S, hpbf, NP_, dp, dppad, hgbf, NG_, dg, dgpad,
            (dg == DH) ? hg_cur : nullptr,
            gp_dst, gp_src, E_GP, P, hg, discat, 512, (long)l * DH);

    // ProteinFusionAttention
    fuse_kernel<<<dim3((unsigned)((NP_ + 3) / 4)), 128, 0, stream>>>(
        pfd, pfs, hp, NP_);

    hp_cur = hp; dp = DH; dppad = DH;
    hd_cur = hd; dd = DH; ddpad = DH;
    hg_cur = hg; dg = DH; dgpad = DH;
  }

  // ---- MLP over treat edges ----
  // x1 = relu(bn(gather(cat) @ W1 + b1))  — fused gather GEMM
  cvt_t(stream, W1, S.wbuf, 1024, 512, 1024);
  {
    dim3 grid((unsigned)(E_T / 16), 4);  // N = 512
    gemm_gather_kernel<<<grid, 128, 0, stream>>>(drugcat, discat, t_src, t_dst,
                                                 S.wbuf, x1, b1);
  }
  bn_stats_kernel<<<512, 256, 0, stream>>>(x1, E_T, 512, bmean, binvstd);
  bn_relu_kernel<<<dim3((unsigned)(((long)E_T * 512 + 255) / 256)), 256, 0,
                   stream>>>(x1, (long)E_T * 512, 512, bmean, binvstd, g1, bn1);

  // x2 = relu(bn(x1 @ W2 + b2))
  cvt_t(stream, W2, S.wbuf, 512, 256, 512);
  gemm(stream, x1, 512, S.wbuf, 512, x2, 256, E_T, 256, 512, b2, 1);
  bn_stats_kernel<<<256, 256, 0, stream>>>(x2, E_T, 256, bmean, binvstd);
  bn_relu_kernel<<<dim3((unsigned)(((long)E_T * 256 + 255) / 256)), 256, 0,
                   stream>>>(x2, (long)E_T * 256, 256, bmean, binvstd, g2, bn2);

  // out = sigmoid(x2 @ W3 + b3)
  final_kernel<<<dim3((unsigned)((E_T + 3) / 4)), 128, 0, stream>>>(
      x2, W3, b3, (float*)d_out, E_T);
}
